// GlobalAttention_7722351198771
// MI455X (gfx1250) — matrix operand order repaired and checked by reading, not compiled, not run on hardware
//
#include <hip/hip_runtime.h>
#include <hip/hip_bf16.h>
#include <stdint.h>

// ---------------- constants ----------------
#define CDIM   256          // hidden / output channels
#define FDIM   256          // input features (== CDIM here)
#define LDE    264          // padded LDS row length (elements) -> 528B rows, 16B aligned
#define CHUNK  32           // node rows processed per iteration
#define THREADS 512         // 16 waves of 32
#define NWAVE  16

typedef __attribute__((ext_vector_type(16))) __bf16 v16bf;
typedef __attribute__((ext_vector_type(8)))  float  v8f;

union Frag { v16bf bf; uint4 q[2]; };

__device__ __forceinline__ unsigned short f2bf(float f) {
  unsigned int u = __float_as_uint(f);
  u += 0x7FFFu + ((u >> 16) & 1u);      // round-to-nearest-even
  return (unsigned short)(u >> 16);
}
__device__ __forceinline__ float gelu_exact(float x) {
  return 0.5f * x * (1.0f + erff(x * 0.70710678118654752f));
}
// A-fragment gather in this code's contiguous k order (k = 16*half + s, matching B) (two contiguous 16B chunks)
__device__ __forceinline__ void load_afrag(Frag& a, const unsigned short* base) {
  a.q[0] = *(const uint4*)(base);
  a.q[1] = *(const uint4*)(base + 8);
}

// -------- tiny helper kernels (negligible work) --------

// transpose+convert node-MLP weights to bf16: Wt[n][k] = bf16(W[k][n])
__global__ void conv_wt_kernel(const float* __restrict__ W1, const float* __restrict__ W2,
                               unsigned short* __restrict__ W1t, unsigned short* __restrict__ W2t) {
  int idx = blockIdx.x * blockDim.x + threadIdx.x;
  if (idx >= CDIM * CDIM) return;
  int n = idx >> 8, k = idx & 255;
  W1t[n * CDIM + k] = f2bf(W1[k * CDIM + n]);
  W2t[n * CDIM + k] = f2bf(W2[k * CDIM + n]);
}

// small dense layer: out[r,c] = (gelu?)(in[r,:] @ W[:,c] + b[c]); rows*256 threads
__global__ void mlp_small_kernel(const float* __restrict__ in, const float* __restrict__ W,
                                 const float* __restrict__ bias, float* __restrict__ outp,
                                 int rows, int apply_gelu) {
  int idx = blockIdx.x * blockDim.x + threadIdx.x;
  if (idx >= rows * CDIM) return;
  int r = idx >> 8, c = idx & 255;
  const float* ip = in + (size_t)r * CDIM;
  float s = bias[c];
  #pragma unroll 4
  for (int k = 0; k < CDIM; ++k) s = fmaf(ip[k], W[k * CDIM + c], s);
  if (apply_gelu) s = gelu_exact(s);
  outp[idx] = s;
}

// -------- fused node-MLP + gate + online segment softmax + pooling --------
// one workgroup per graph (batch sorted -> contiguous node range).
// Weight B-fragments live in VGPRs; A-frags double-buffered; two independent
// accumulator chains so WMMAs have no back-to-back RAW hazard.
__global__ __launch_bounds__(THREADS) void ga_fused_kernel(
    const float* __restrict__ x, const int* __restrict__ batch,
    const unsigned short* __restrict__ w1t, const unsigned short* __restrict__ w2t,
    const float* __restrict__ b1, const float* __restrict__ b2,
    const float* __restrict__ qmat, float* __restrict__ out, int N) {
  __shared__ unsigned short sXT[CHUNK * LDE];   // x chunk, bf16
  __shared__ unsigned short sHT[CHUNK * LDE];   // hidden chunk, bf16
  __shared__ float sGacc[CHUNK];
  __shared__ float sGw[CHUNK];
  __shared__ float sMisc[4];                    // 0=running max, 1=running sum, 2=scale
  __shared__ int   sSeg[2];

  const int b   = blockIdx.x;
  const int tid = threadIdx.x;
  const int wave = tid >> 5, lane = tid & 31;
  const int half = lane >> 4, ln = lane & 15;
  const int n0 = wave * 16;                     // this wave's 16 output columns
  const int col = n0 + ln;

  // ---- per-lane loop invariants from global (L2-resident) ----
  const float qv  = qmat[(size_t)b * CDIM + col] * 0.0625f;  // fold 1/sqrt(C)=1/16
  const float b1v = b1[col];
  const float b2v = b2[col];

  // ---- preload all weight B-fragments into registers (constant per wave) ----
  v16bf wf1[8], wf2[8];
  #pragma unroll
  for (int ks = 0; ks < 8; ++ks) {
    Frag f;
    const unsigned short* p1 = w1t + col * CDIM + ks * 32 + half * 16;
    f.q[0] = *(const uint4*)(p1);
    f.q[1] = *(const uint4*)(p1 + 8);
    wf1[ks] = f.bf;
    const unsigned short* p2 = w2t + col * CDIM + ks * 32 + half * 16;
    f.q[0] = *(const uint4*)(p2);
    f.q[1] = *(const uint4*)(p2 + 8);
    wf2[ks] = f.bf;
  }

  if (tid < CHUNK) sGacc[tid] = 0.0f;
  if (tid == 0) {
    // binary search segment bounds in sorted batch
    int lo = 0, hi = N;
    while (lo < hi) { int mid = (lo + hi) >> 1; if (batch[mid] < b) lo = mid + 1; else hi = mid; }
    int s = lo;
    lo = s; hi = N;
    while (lo < hi) { int mid = (lo + hi) >> 1; if (batch[mid] < b + 1) lo = mid + 1; else hi = mid; }
    sSeg[0] = s; sSeg[1] = lo;
    sMisc[0] = -3.0e38f;   // running max
    sMisc[1] = 0.0f;       // running sum
    sMisc[2] = 0.0f;       // scale
  }
  __syncthreads();
  const int seg_s = sSeg[0], seg_e = sSeg[1];

  float accL = 0.0f;                            // per-lane slice of column accumulator
  const float4* x4 = (const float4*)x;

  // per-lane A-row bases within a chunk tile (tile0 row = ln, tile1 row = ln+16)
  const int aoff0 = (ln)      * LDE + half * 16;
  const int aoff1 = (ln + 16) * LDE + half * 16;

  for (int rs = seg_s; rs < seg_e; rs += CHUNK) {
    // ---- load 32 x-rows, convert fp32->bf16 into LDS ----
    #pragma unroll
    for (int i = 0; i < 4; ++i) {
      int idx  = tid + i * THREADS;             // 2048 float4 per chunk
      int row  = idx >> 6, c4 = idx & 63;
      int node = rs + row;
      float4 v = make_float4(0.f, 0.f, 0.f, 0.f);
      if (node < seg_e) {
        v = x4[(size_t)node * 64 + c4];
        if (node + CHUNK < seg_e)               // prefetch next chunk (global_prefetch_b8)
          __builtin_prefetch(&x4[(size_t)(node + CHUNK) * 64 + c4], 0, 1);
      }
      uint2 p;
      p.x = (unsigned)f2bf(v.x) | ((unsigned)f2bf(v.y) << 16);
      p.y = (unsigned)f2bf(v.z) | ((unsigned)f2bf(v.w) << 16);
      *(uint2*)(sXT + row * LDE + c4 * 4) = p;
    }
    __syncthreads();

    // ---- GEMM1: h = gelu(x @ W1 + b1); ks outer, both tiles inner, 2-deep A pipeline ----
    {
      v8f acc0 = {0.f,0.f,0.f,0.f,0.f,0.f,0.f,0.f};
      v8f acc1 = {0.f,0.f,0.f,0.f,0.f,0.f,0.f,0.f};
      Frag ca0, ca1, na0, na1;
      load_afrag(ca0, sXT + aoff0);
      load_afrag(ca1, sXT + aoff1);
      #pragma unroll
      for (int ks = 0; ks < 8; ++ks) {
        if (ks < 7) {                           // issue next-k loads before this-k WMMAs
          load_afrag(na0, sXT + aoff0 + (ks + 1) * 32);
          load_afrag(na1, sXT + aoff1 + (ks + 1) * 32);
        }
        acc0 = __builtin_amdgcn_wmma_f32_16x16x32_bf16(false, ca0.bf, false, wf1[ks],
                                                       (short)0, acc0, false, false);
        acc1 = __builtin_amdgcn_wmma_f32_16x16x32_bf16(false, ca1.bf, false, wf1[ks],
                                                       (short)0, acc1, false, false);
        ca0 = na0; ca1 = na1;
      }
      #pragma unroll
      for (int r = 0; r < 8; ++r) {
        int mr0 = r + half * 8;                 // C/D layout: VGPR r -> M = r (+8 hi half)
        sHT[mr0 * LDE + col]        = f2bf(gelu_exact(acc0[r] + b1v));
        sHT[(mr0 + 16) * LDE + col] = f2bf(gelu_exact(acc1[r] + b1v));
      }
    }
    __syncthreads();

    // ---- GEMM2: xn = h @ W2 + b2 (kept in regs) + gate partial dot ----
    v8f xacc0 = {0.f,0.f,0.f,0.f,0.f,0.f,0.f,0.f};
    v8f xacc1 = {0.f,0.f,0.f,0.f,0.f,0.f,0.f,0.f};
    {
      Frag ca0, ca1, na0, na1;
      load_afrag(ca0, sHT + aoff0);
      load_afrag(ca1, sHT + aoff1);
      #pragma unroll
      for (int ks = 0; ks < 8; ++ks) {
        if (ks < 7) {
          load_afrag(na0, sHT + aoff0 + (ks + 1) * 32);
          load_afrag(na1, sHT + aoff1 + (ks + 1) * 32);
        }
        xacc0 = __builtin_amdgcn_wmma_f32_16x16x32_bf16(false, ca0.bf, false, wf2[ks],
                                                        (short)0, xacc0, false, false);
        xacc1 = __builtin_amdgcn_wmma_f32_16x16x32_bf16(false, ca1.bf, false, wf2[ks],
                                                        (short)0, xacc1, false, false);
        ca0 = na0; ca1 = na1;
      }
    }
    #pragma unroll
    for (int r = 0; r < 8; ++r) {
      float xv0 = xacc0[r] + b2v;
      float xv1 = xacc1[r] + b2v;
      xacc0[r] = xv0;
      xacc1[r] = xv1;
      float p0 = xv0 * qv;                      // gate partial: xn[m,n]*q[n]/16
      float p1 = xv1 * qv;
      p0 += __shfl_xor(p0, 1, 32);
      p1 += __shfl_xor(p1, 1, 32);
      p0 += __shfl_xor(p0, 2, 32);
      p1 += __shfl_xor(p1, 2, 32);
      p0 += __shfl_xor(p0, 4, 32);
      p1 += __shfl_xor(p1, 4, 32);
      p0 += __shfl_xor(p0, 8, 32);
      p1 += __shfl_xor(p1, 8, 32);
      if (ln == 0) {
        atomicAdd(&sGacc[r + half * 8], p0);
        atomicAdd(&sGacc[16 + r + half * 8], p1);
      }
    }
    __syncthreads();

    // ---- online softmax state update (wave 0) ----
    if (wave == 0) {
      float mrun = sMisc[0], srun = sMisc[1];
      float g = ((rs + lane) < seg_e) ? sGacc[lane] : -3.0e38f;
      float cmax = g;
      cmax = fmaxf(cmax, __shfl_xor(cmax, 1, 32));
      cmax = fmaxf(cmax, __shfl_xor(cmax, 2, 32));
      cmax = fmaxf(cmax, __shfl_xor(cmax, 4, 32));
      cmax = fmaxf(cmax, __shfl_xor(cmax, 8, 32));
      cmax = fmaxf(cmax, __shfl_xor(cmax, 16, 32));
      float newm = fmaxf(mrun, cmax);
      float sc = __expf(mrun - newm);
      float w  = __expf(g - newm);              // invalid rows underflow to 0
      float cs = w;
      cs += __shfl_xor(cs, 1, 32);
      cs += __shfl_xor(cs, 2, 32);
      cs += __shfl_xor(cs, 4, 32);
      cs += __shfl_xor(cs, 8, 32);
      cs += __shfl_xor(cs, 16, 32);
      sGw[lane] = w;
      if (lane == 0) { sMisc[0] = newm; sMisc[1] = srun * sc + cs; sMisc[2] = sc; }
    }
    __syncthreads();

    // ---- rescale accumulator and add weighted rows ----
    float sc = sMisc[2];
    accL *= sc;
    #pragma unroll
    for (int r = 0; r < 8; ++r) {
      accL = fmaf(sGw[r + half * 8],      xacc0[r], accL);
      accL = fmaf(sGw[16 + r + half * 8], xacc1[r], accL);
    }
    if (tid < CHUNK) sGacc[tid] = 0.0f;         // ready for next chunk
  }

  // ---- combine halves (rows 0-7 slice + rows 8-15 slice) and normalize ----
  float tot = accL + __shfl_xor(accL, 16, 32);
  if (half == 0)
    out[(size_t)b * CDIM + col] = tot / (sMisc[1] + 1e-16f);
}

// ---------------- host launcher ----------------
extern "C" void kernel_launch(void* const* d_in, const int* in_sizes, int n_in,
                              void* d_out, int out_size, void* d_ws, size_t ws_size,
                              hipStream_t stream) {
  const float* x     = (const float*)d_in[0];
  const float* u     = (const float*)d_in[1];
  const int*   batch = (const int*)d_in[2];
  // d_in[3] = size scalar (== B, derived from u instead)
  const float* Wn1 = (const float*)d_in[4];
  const float* bn1 = (const float*)d_in[5];
  const float* Wn2 = (const float*)d_in[6];
  const float* bn2 = (const float*)d_in[7];
  const float* Wq1 = (const float*)d_in[8];
  const float* bq1 = (const float*)d_in[9];
  const float* Wq2 = (const float*)d_in[10];
  const float* bq2 = (const float*)d_in[11];

  const int N = in_sizes[0] / FDIM;
  const int B = in_sizes[1] / CDIM;
  float* outp = (float*)d_out;

  // workspace: [W1t bf16 128KB][W2t bf16 128KB][hq f32 B*C][q f32 B*C]
  char* ws = (char*)d_ws;
  unsigned short* W1t = (unsigned short*)ws;
  unsigned short* W2t = (unsigned short*)(ws + (size_t)CDIM * CDIM * 2);
  float* hq = (float*)(ws + (size_t)CDIM * CDIM * 4);
  float* qm = (float*)(ws + (size_t)CDIM * CDIM * 4 + (size_t)B * CDIM * 4);

  conv_wt_kernel<<<(CDIM * CDIM + 255) / 256, 256, 0, stream>>>(Wn1, Wn2, W1t, W2t);
  mlp_small_kernel<<<(B * CDIM + 255) / 256, 256, 0, stream>>>(u, Wq1, bq1, hq, B, 1);
  mlp_small_kernel<<<(B * CDIM + 255) / 256, 256, 0, stream>>>(hq, Wq2, bq2, qm, B, 0);

  ga_fused_kernel<<<B, THREADS, 0, stream>>>(x, batch, W1t, W2t, bn1, bn2, qm, outp, N);
}
